// pool_7275674600515
// MI455X (gfx1250) — compile-verified, matrix operand order checked by reading, not run on hardware
//
#include <hip/hip_runtime.h>
#include <hip/hip_bf16.h>

typedef __bf16 bf16_t;
typedef __attribute__((ext_vector_type(8)))  __bf16 v8bf;
typedef __attribute__((ext_vector_type(16))) __bf16 v16bf;
typedef __attribute__((ext_vector_type(8)))  float  v8f;

// problem dims (compile-time so all %, / fold to masks/shifts)
constexpr unsigned GB = 4, GH = 128, GW = 128, GDIM = 256, GMID = 128;
constexpr unsigned GNPIX = GB * GH * GW;

#define DEV static __device__ __forceinline__

DEV v16bf cat8(v8bf lo, v8bf hi) {
    v16bf r;
#pragma unroll
    for (int i = 0; i < 8; ++i) { r[i] = lo[i]; r[i + 8] = hi[i]; }
    return r;
}

// wait for all of this wave's outstanding async global->LDS transfers
DEV void wait_async0() { asm volatile("s_wait_asynccnt 0" ::: "memory"); }

// issue one 16B async global->LDS copy (GV mode, per-lane addresses).
// LDS destination offset = low 32 bits of the generic shared pointer
// (flat->LDS mapping truncates to addr[31:0]).
DEV void async_copy16(const bf16_t* gsrc, bf16_t* ldst)
{
    const unsigned dstoff = (unsigned)(uintptr_t)ldst;
    const unsigned long long g = (unsigned long long)(uintptr_t)gsrc;
    asm volatile("global_load_async_to_lds_b128 %0, %1, off"
                 :: "v"(dstoff), "v"(g) : "memory");
}

// ---------------------------------------------------------------------------
// Cooperative async stage of one (dy,dx) tap: PIXB pixels x C channels -> LDS.
// 256 threads issue 16B async copies, zero-filling the pad-1 halo with plain
// LDS stores. Uniform counted loop (TOT is a multiple of 256) and unsigned
// index math -> no EXEC-mask loop control, no signed-div fixups.
// LDS row stride = STRIDE elements (16B pad) -> pixel rows advance 4 banks,
// so 16 lanes doing b128 reads of one channel chunk are bank-conflict free.
// ---------------------------------------------------------------------------
template <unsigned C, unsigned STRIDE, unsigned PIXB>
DEV void stage_tap_async(const bf16_t* __restrict__ act, bf16_t* buf,
                         unsigned n_block, int dy, int dx)
{
    constexpr unsigned CH   = C / 8;       // 16B chunks per pixel
    constexpr unsigned ITER = PIXB * CH / 256;
#pragma unroll 1
    for (unsigned it = 0; it < ITER; ++it) {
        const unsigned i     = it * 256 + threadIdx.x;
        const unsigned pixel = i / CH;
        const unsigned cc    = i % CH;
        const unsigned n = n_block + pixel;
        const unsigned w = n % GW;
        const unsigned h = (n / GW) % GH;
        const unsigned b = n / (GW * GH);
        const int hy = (int)h + dy, wx = (int)w + dx;
        bf16_t* dst = buf + pixel * STRIDE + cc * 8;
        if (hy >= 0 && hy < (int)GH && wx >= 0 && wx < (int)GW) {
            async_copy16(act + ((size_t)((b * GH + hy) * GW + wx) * C + cc * 8), dst);
        } else {
            v8bf z;
#pragma unroll
            for (int j = 0; j < 8; ++j) z[j] = (__bf16)0.0f;
            *(v8bf*)dst = z;
        }
    }
}

// ---------------------------------------------------------------------------
// Implicit-GEMM 3x3 conv (pad 1) with folded BN, optional fused 1x1 second
// GEMM, WMMA bf16 -> f32.  M = Cout, N = 128 pixels/block, K = 9*CIN (+CIN2).
// Wave tile: 16 couts x 128 pixels (8 accumulators, A-frag reuse x8);
// 8 waves/block cover 128 couts; gridDim.y covers NCOUT/128.
// Activations double-buffered in LDS via async global->LDS copies: tap k+1
// streams in while tap k feeds the WMMA pipe. Weights stream from global (L2).
// OUTMODE 0: bf16 NHWC intermediate.  OUTMODE 1: f32 NCHW final output.
// ---------------------------------------------------------------------------
template <unsigned CIN, unsigned CIN2, unsigned NCOUT, bool RELU, int OUTMODE>
__global__ __launch_bounds__(256) void conv_wmma(
    const bf16_t* __restrict__ act,   // NHWC bf16 [B,H,W,CIN]
    const bf16_t* __restrict__ wgt,   // [NCOUT, 9*CIN] (tap-major, cin-minor)
    const float*  __restrict__ bias,  // [NCOUT]
    const bf16_t* __restrict__ act2,  // NHWC bf16 [B,H,W,CIN2] (1x1 path)
    const bf16_t* __restrict__ wgt2,  // [NCOUT, CIN2]
    const float*  __restrict__ bias2, // [NCOUT]
    void* __restrict__ outp)
{
    constexpr unsigned NT     = 8;            // N-tiles (of 16 pixels) per wave
    constexpr unsigned PIXB   = NT * 16;      // 128 pixels per block
    constexpr unsigned CMAX   = (CIN2 > CIN ? CIN2 : CIN);
    constexpr unsigned STRIDE = CMAX + 8;     // LDS row stride (elements)
    __shared__ bf16_t smem[2 * PIXB * STRIDE];

    const unsigned lane = threadIdx.x & 31;
    const unsigned wave = threadIdx.x >> 5;
    const unsigned l15  = lane & 15;
    const unsigned koff = (lane & 16) ? 8 : 0;    // K half per ISA A/B layout
    const unsigned cout_base = (blockIdx.y * 8 + wave) * 16;
    const unsigned cout_row  = cout_base + l15;
    const unsigned n_block   = blockIdx.x * PIXB;

    unsigned rowb[NT];                            // LDS row base per N-tile
#pragma unroll
    for (unsigned t = 0; t < NT; ++t)
        rowb[t] = (t * 16 + l15) * STRIDE + koff;

    v8f acc[NT];
#pragma unroll
    for (unsigned t = 0; t < NT; ++t)
#pragma unroll
        for (int i = 0; i < 8; ++i) acc[t][i] = 0.0f;

    const bf16_t* wrow = wgt + (size_t)cout_row * (9 * CIN);
    bf16_t* const buf0 = smem;
    bf16_t* const buf1 = smem + PIXB * STRIDE;

    // prologue: stage tap 0
    stage_tap_async<CIN, STRIDE, PIXB>(act, buf0, n_block, -1, -1);
    wait_async0();
    __syncthreads();

#pragma unroll 1
    for (unsigned kyx = 0; kyx < 9; ++kyx) {
        bf16_t* const cur = (kyx & 1) ? buf1 : buf0;
        bf16_t* const nxt = (kyx & 1) ? buf0 : buf1;

        // overlap: stream next tap (or the 1x1 slab) while computing this one
        if (kyx < 8) {
            const unsigned nd = kyx + 1;
            stage_tap_async<CIN, STRIDE, PIXB>(act, nxt, n_block,
                                               (int)(nd / 3) - 1, (int)(nd % 3) - 1);
        } else if constexpr (CIN2 > 0) {
            stage_tap_async<CIN2, STRIDE, PIXB>(act2, nxt, n_block, 0, 0);
        }

        const bf16_t* wk = wrow + kyx * CIN;
        if (kyx < 8) __builtin_prefetch(wk + CIN, 0, 0);   // next-tap weights
#pragma unroll 1
        for (unsigned cb = 0; cb < CIN / 32; ++cb) {
            const unsigned cin0 = cb * 32;
            v16bf afrag = cat8(*(const v8bf*)(wk + cin0 + koff),
                               *(const v8bf*)(wk + cin0 + 16 + koff));
#pragma unroll
            for (unsigned t = 0; t < NT; ++t) {
                const bf16_t* p = cur + rowb[t] + cin0;
                v16bf bfrag = cat8(*(const v8bf*)p, *(const v8bf*)(p + 16));
                acc[t] = __builtin_amdgcn_wmma_f32_16x16x32_bf16(
                    false, afrag, false, bfrag, (short)0, acc[t], false, false);
            }
        }
        wait_async0();        // next-tap async copies have landed in LDS
        __syncthreads();      // all waves done reading cur / staging nxt
    }

    if constexpr (CIN2 > 0) {                // fused 1x1 skip GEMM (tap "9")
        bf16_t* const cur = buf1;            // 9 & 1
        const bf16_t* wrow2 = wgt2 + (size_t)cout_row * CIN2;
#pragma unroll 1
        for (unsigned cb = 0; cb < CIN2 / 32; ++cb) {
            const unsigned cin0 = cb * 32;
            v16bf afrag = cat8(*(const v8bf*)(wrow2 + cin0 + koff),
                               *(const v8bf*)(wrow2 + cin0 + 16 + koff));
#pragma unroll
            for (unsigned t = 0; t < NT; ++t) {
                const bf16_t* p = cur + rowb[t] + cin0;
                v16bf bfrag = cat8(*(const v8bf*)p, *(const v8bf*)(p + 16));
                acc[t] = __builtin_amdgcn_wmma_f32_16x16x32_bf16(
                    false, afrag, false, bfrag, (short)0, acc[t], false, false);
            }
        }
    }

    // D layout: VGPR v -> cout = cout_base + v + (lane>=16 ? 8 : 0); N = lane&15
    const unsigned ch0 = cout_base + ((lane & 16) ? 8 : 0);
    float bv[8];
#pragma unroll
    for (int v = 0; v < 8; ++v) {
        float b0 = bias[ch0 + v];
        if constexpr (CIN2 > 0) b0 += bias2[ch0 + v];
        bv[v] = b0;
    }

    if constexpr (OUTMODE == 0) {
        bf16_t* outb = (bf16_t*)outp;
#pragma unroll
        for (unsigned t = 0; t < NT; ++t) {
            const unsigned n = n_block + t * 16 + l15;
            v8bf o;
#pragma unroll
            for (int v = 0; v < 8; ++v) {
                float r = acc[t][v] + bv[v];
                if (RELU) r = fmaxf(r, 0.0f);
                o[v] = (__bf16)r;
            }
            *(v8bf*)(outb + (size_t)n * NCOUT + ch0) = o;
        }
    } else {
        float* outf = (float*)outp;
        constexpr unsigned HW = GH * GW;
#pragma unroll
        for (unsigned t = 0; t < NT; ++t) {
            const unsigned n  = n_block + t * 16 + l15;
            const unsigned b  = n / HW;
            const unsigned hw = n % HW;
#pragma unroll
            for (int v = 0; v < 8; ++v) {
                float r = acc[t][v] + bv[v];
                if (RELU) r = fmaxf(r, 0.0f);
                outf[(size_t)(b * NCOUT + ch0 + v) * HW + hw] = r;
            }
        }
    }
}

// ------------------------- data-prep kernels -------------------------------

// NCHW f32 -> NHWC bf16 (x: C = GDIM)
__global__ void to_nhwc_bf16(const float* __restrict__ x, bf16_t* __restrict__ out)
{
    unsigned idx = blockIdx.x * blockDim.x + threadIdx.x;
    constexpr unsigned total = GNPIX * GDIM;
    if (idx >= total) return;
    unsigned w = idx % GW; unsigned t = idx / GW;
    unsigned h = t % GH; t /= GH;
    unsigned c = t % GDIM; unsigned b = t / GDIM;
    out[(((size_t)b * GH + h) * GW + w) * GDIM + c] = (__bf16)x[idx];
}

// Fold BN scale into 3x3 weights: [Cout,Cin,3,3] f32 -> [Cout][(ky*3+kx)*Cin+cin] bf16
__global__ void prep_w3(const float* __restrict__ w, const float* __restrict__ g,
                        const float* __restrict__ b, const float* __restrict__ m,
                        const float* __restrict__ v,
                        bf16_t* __restrict__ wout, float* __restrict__ bout,
                        unsigned Cout, unsigned Cin)
{
    unsigned idx = blockIdx.x * blockDim.x + threadIdx.x;
    unsigned total = Cout * Cin * 9;
    if (idx >= total) return;
    unsigned kx = idx % 3; unsigned t = idx / 3;
    unsigned ky = t % 3;   t /= 3;
    unsigned cin = t % Cin; unsigned cout = t / Cin;
    float s = g[cout] * rsqrtf(v[cout] + 1e-5f);
    wout[(size_t)cout * (9 * Cin) + (ky * 3 + kx) * Cin + cin] = (__bf16)(w[idx] * s);
    if (cin == 0 && ky == 0 && kx == 0) bout[cout] = b[cout] - m[cout] * s;
}

// Fold BN scale into 1x1 weights: [Cout,Cin,1,1] f32 -> [Cout,Cin] bf16
__global__ void prep_w1(const float* __restrict__ w, const float* __restrict__ g,
                        const float* __restrict__ b, const float* __restrict__ m,
                        const float* __restrict__ v,
                        bf16_t* __restrict__ wout, float* __restrict__ bout,
                        unsigned Cout, unsigned Cin)
{
    unsigned idx = blockIdx.x * blockDim.x + threadIdx.x;
    if (idx >= Cout * Cin) return;
    unsigned cin = idx % Cin, cout = idx / Cin;
    float s = g[cout] * rsqrtf(v[cout] + 1e-5f);
    wout[idx] = (__bf16)(w[idx] * s);
    if (cin == 0) bout[cout] = b[cout] - m[cout] * s;
}

// TopPool: reverse cummax over H; writes tl. One thread per (b,w,c) line.
__global__ void toppool(const bf16_t* __restrict__ p1, bf16_t* __restrict__ tl)
{
    unsigned idx = blockIdx.x * blockDim.x + threadIdx.x;
    if (idx >= GB * GW * GMID) return;
    unsigned c = idx % GMID; unsigned t = idx / GMID;
    unsigned w = t % GW;     unsigned b = t / GW;
    float run = -3.0e38f;
    for (int h = GH - 1; h >= 0; --h) {
        size_t off = (((size_t)(b * GH + h) * GW + w) * GMID + c);
        run = fmaxf(run, (float)p1[off]);
        tl[off] = (__bf16)run;
    }
}

// LeftPool: reverse cummax over W; tl += left. One thread per (b,h,c) line.
__global__ void leftpool_add(const bf16_t* __restrict__ p2, bf16_t* __restrict__ tl)
{
    unsigned idx = blockIdx.x * blockDim.x + threadIdx.x;
    if (idx >= GB * GH * GMID) return;
    unsigned c = idx % GMID; unsigned t = idx / GMID;
    unsigned h = t % GH;     unsigned b = t / GH;
    float run = -3.0e38f;
    for (int w = GW - 1; w >= 0; --w) {
        size_t off = (((size_t)(b * GH + h) * GW + w) * GMID + c);
        run = fmaxf(run, (float)p2[off]);
        tl[off] = (__bf16)((float)tl[off] + run);
    }
}

// ---------------------------------------------------------------------------

extern "C" void kernel_launch(void* const* d_in, const int* in_sizes, int n_in,
                              void* d_out, int out_size, void* d_ws, size_t ws_size,
                              hipStream_t stream)
{
    constexpr unsigned B = GB, DIM = GDIM, MID = GMID, H = GH, W = GW;
    constexpr unsigned NPIX = GNPIX;
    constexpr unsigned PIXB = 128;

    const float* x    = (const float*)d_in[0];
    const float* w_p1 = (const float*)d_in[1];
    const float* g_p1 = (const float*)d_in[2];
    const float* b_p1 = (const float*)d_in[3];
    const float* m_p1 = (const float*)d_in[4];
    const float* v_p1 = (const float*)d_in[5];
    const float* w_p2 = (const float*)d_in[6];
    const float* g_p2 = (const float*)d_in[7];
    const float* b_p2 = (const float*)d_in[8];
    const float* m_p2 = (const float*)d_in[9];
    const float* v_p2 = (const float*)d_in[10];
    const float* w_pc = (const float*)d_in[11];
    const float* g_pb = (const float*)d_in[12];
    const float* b_pb = (const float*)d_in[13];
    const float* m_pb = (const float*)d_in[14];
    const float* v_pb = (const float*)d_in[15];
    const float* w_c1 = (const float*)d_in[16];
    const float* g_b1 = (const float*)d_in[17];
    const float* b_b1 = (const float*)d_in[18];
    const float* m_b1 = (const float*)d_in[19];
    const float* v_b1 = (const float*)d_in[20];
    const float* w_c2 = (const float*)d_in[21];
    const float* g_c2 = (const float*)d_in[22];
    const float* b_c2 = (const float*)d_in[23];
    const float* m_c2 = (const float*)d_in[24];
    const float* v_c2 = (const float*)d_in[25];

    // workspace carve-out (256B aligned)
    char* ws = (char*)d_ws;
    size_t off = 0;
    auto alloc = [&](size_t bytes) -> void* {
        void* p = ws + off;
        off += (bytes + 255) & ~(size_t)255;
        return p;
    };
    bf16_t* xbf   = (bf16_t*)alloc((size_t)NPIX * DIM * 2);   // x NHWC bf16
    bf16_t* p1b   = (bf16_t*)alloc((size_t)NPIX * MID * 2);
    bf16_t* p2b   = (bf16_t*)alloc((size_t)NPIX * MID * 2);
    bf16_t* tl    = (bf16_t*)alloc((size_t)NPIX * MID * 2);   // top+left
    bf16_t* relu1 = (bf16_t*)alloc((size_t)NPIX * DIM * 2);
    bf16_t* wp1s  = (bf16_t*)alloc((size_t)MID * 9 * DIM * 2);
    float*  bp1   = (float*)alloc(MID * 4);
    bf16_t* wp2s  = (bf16_t*)alloc((size_t)MID * 9 * DIM * 2);
    float*  bp2   = (float*)alloc(MID * 4);
    bf16_t* wpcs  = (bf16_t*)alloc((size_t)DIM * 9 * MID * 2);
    float*  bpc   = (float*)alloc(DIM * 4);
    bf16_t* wc1s  = (bf16_t*)alloc((size_t)DIM * DIM * 2);
    float*  bc1   = (float*)alloc(DIM * 4);
    bf16_t* wc2s  = (bf16_t*)alloc((size_t)DIM * 9 * DIM * 2);
    float*  bc2   = (float*)alloc(DIM * 4);

    // ---- prep: layout conversion + BN folding ----
    to_nhwc_bf16<<<(NPIX * DIM + 255) / 256, 256, 0, stream>>>(x, xbf);
    prep_w3<<<(MID * DIM * 9 + 255) / 256, 256, 0, stream>>>(
        w_p1, g_p1, b_p1, m_p1, v_p1, wp1s, bp1, MID, DIM);
    prep_w3<<<(MID * DIM * 9 + 255) / 256, 256, 0, stream>>>(
        w_p2, g_p2, b_p2, m_p2, v_p2, wp2s, bp2, MID, DIM);
    prep_w3<<<(DIM * MID * 9 + 255) / 256, 256, 0, stream>>>(
        w_pc, g_pb, b_pb, m_pb, v_pb, wpcs, bpc, DIM, MID);
    prep_w1<<<(DIM * DIM + 255) / 256, 256, 0, stream>>>(
        w_c1, g_b1, b_b1, m_b1, v_b1, wc1s, bc1, DIM, DIM);
    prep_w3<<<(DIM * DIM * 9 + 255) / 256, 256, 0, stream>>>(
        w_c2, g_c2, b_c2, m_c2, v_c2, wc2s, bc2, DIM, DIM);

    // ---- p1, p2: conv3x3(256->128)+BN+ReLU via WMMA ----
    {
        dim3 grid(NPIX / PIXB, MID / 128);
        conv_wmma<DIM, 0, MID, true, 0><<<grid, 256, 0, stream>>>(
            xbf, wp1s, bp1, nullptr, nullptr, nullptr, (void*)p1b);
        conv_wmma<DIM, 0, MID, true, 0><<<grid, 256, 0, stream>>>(
            xbf, wp2s, bp2, nullptr, nullptr, nullptr, (void*)p2b);
    }

    // ---- corner pools: tl = revcummax_H(p1) + revcummax_W(p2) ----
    toppool<<<(B * W * MID + 255) / 256, 256, 0, stream>>>(p1b, tl);
    leftpool_add<<<(B * H * MID + 255) / 256, 256, 0, stream>>>(p2b, tl);

    // ---- fused: relu( conv3x3(tl,128->256)+BN + conv1x1(x,256->256)+BN ) ----
    {
        dim3 grid(NPIX / PIXB, DIM / 128);
        conv_wmma<MID, DIM, DIM, true, 0><<<grid, 256, 0, stream>>>(
            tl, wpcs, bpc, xbf, wc1s, bc1, (void*)relu1);
    }

    // ---- final: conv3x3(256->256)+BN+ReLU -> f32 NCHW d_out ----
    {
        dim3 grid(NPIX / PIXB, DIM / 128);
        conv_wmma<DIM, 0, DIM, true, 1><<<grid, 256, 0, stream>>>(
            relu1, wc2s, bc2, nullptr, nullptr, nullptr, d_out);
    }
}
